// SelfAttention_86569360818603
// MI455X (gfx1250) — matrix operand order repaired and checked by reading, not compiled, not run on hardware
//
#include <hip/hip_runtime.h>

// ---------------------------------------------------------------------------
// Self-attention for MI455X (gfx1250), wave32 + v_wmma_f32_16x16x32_bf16,
// double-buffered LDS pipeline, async global->LDS (ASYNCcnt) for bf16 copies,
// v_perm_b32 packed fp32->bf16 conversion.
//   1) QKV projection GEMM (fp32 -> bf16 in workspace)
//   2) scores = Q K^T * 1/32 (bf16 in, fp32 out)   [fully async-staged]
//   3) row softmax (fp32 -> bf16 P overlaid in score buffer)
//   4) out = P V (bf16 in, fp32 out)
// Workspace: [0,48M) K/Q/V bf16 ; [48M,112M) scores fp32 (P bf16 overlaid)
// ---------------------------------------------------------------------------

typedef __attribute__((ext_vector_type(16))) __bf16 v16bf;
typedef __attribute__((ext_vector_type(8)))  __bf16 v8bf;
typedef __attribute__((ext_vector_type(8)))  float  v8f;

__device__ __forceinline__ unsigned fbits(float f) {
  union { float f; unsigned u; } x; x.f = f; return x.u;
}
// single fp32 -> bf16 (round-half-up): 2 VALU
__device__ __forceinline__ unsigned short f2bf(float f) {
  return (unsigned short)((fbits(f) + 0x8000u) >> 16);
}
// pair fp32 -> packed 2xbf16 via v_perm_b32: 3 VALU per 2 elements
__device__ __forceinline__ unsigned f2bf2(float lo, float hi) {
  // D = {S0.hi16, S1.hi16}  (S1.hi16 in low half)
  return __builtin_amdgcn_perm(fbits(hi) + 0x8000u, fbits(lo) + 0x8000u, 0x07060302u);
}

// 16B async copy global -> LDS (per lane). Tracked by ASYNCcnt.
__device__ __forceinline__ void async_cp16(unsigned lds_off, const void* gaddr) {
  asm volatile("global_load_async_to_lds_b128 %0, %1, off"
               :: "v"(lds_off), "v"(gaddr) : "memory");
}
__device__ __forceinline__ void wait_async0() {
  asm volatile("s_wait_asynccnt 0x0" ::: "memory");
}

union FragU { v16bf v; v8bf h[2]; };

// C[z] = alpha * A[z] * B[z]
//   A: row-major [M][K] (AT elements, lda)
//   B: BTRANS ? global [K][N] (transposed into LDS) : global [N][K] row-major
template <typename AT, typename BT, typename CT, bool BTRANS>
__global__ __launch_bounds__(256) void gemm_wmma(
    const AT* __restrict__ A, long lda, long bsA,
    const BT* __restrict__ B, long ldb, long bsB,
    CT* __restrict__ C, long ldc, long bsC,
    int Kdim, float alpha)
{
  constexpr int BM = 128, BN = 128, BK = 32, LSTR = 40; // 80B rows: conflict-free b128
  constexpr bool ASYNC_A = !__is_same(AT, float);       // bf16 straight copy
  constexpr bool ASYNC_B = !BTRANS;                     // bf16 straight copy
  __shared__ unsigned short sA[2][BM * LSTR];
  __shared__ unsigned short sB[2][BN * LSTR];

  const int tid  = threadIdx.x;
  const int lane = tid & 31;
  const int wv   = tid >> 5;
  const int wm   = wv >> 1;            // 0..3 -> 32-row slice
  const int wn   = wv & 1;             // 0..1 -> 64-col slice
  const int l16  = lane & 15;
  const int hf   = lane >> 4;
  const long m0  = (long)blockIdx.x * BM;
  const long n0  = (long)blockIdx.y * BN;

  A += (long)blockIdx.z * bsA;
  B += (long)blockIdx.z * bsB;
  C += (long)blockIdx.z * bsC;

  // --- fixed per-thread staging coordinates (no dynamic-bound loops) ---
  int aRow[4], aCol[4], bRow[4], bCol[4];
  if constexpr (__is_same(AT, float)) {
#pragma unroll
    for (int i = 0; i < 4; ++i) { int c = tid + 256 * i; aRow[i] = c >> 3; aCol[i] = (c & 7) * 4; }
  } else {
#pragma unroll
    for (int i = 0; i < 2; ++i) { int c = tid + 256 * i; aRow[i] = c >> 2; aCol[i] = (c & 3) * 8; }
  }
  if constexpr (!BTRANS) {
#pragma unroll
    for (int i = 0; i < 2; ++i) { int c = tid + 256 * i; bRow[i] = c >> 2; bCol[i] = (c & 3) * 8; }
  } else if constexpr (__is_same(BT, float)) {
#pragma unroll
    for (int i = 0; i < 4; ++i) { int c = tid + 256 * i; bRow[i] = c >> 5; bCol[i] = (c & 31) * 4; }
  } else {
#pragma unroll
    for (int i = 0; i < 2; ++i) { int c = tid + 256 * i; bRow[i] = c >> 4; bCol[i] = (c & 15) * 8; }
  }

  float4 raf[4], rbf[4];   // fp32 staging regs
  uint4  rbb[2];           // bf16 transpose staging regs

  v8f acc[2][4];
#pragma unroll
  for (int mi = 0; mi < 2; ++mi)
#pragma unroll
    for (int ni = 0; ni < 4; ++ni)
#pragma unroll
      for (int e = 0; e < 8; ++e) acc[mi][ni][e] = 0.0f;

  // ---- staging helpers ----
  auto issueA = [&](int step, int buf) {
    const long kk = (long)step * BK;
    if constexpr (ASYNC_A) {
#pragma unroll
      for (int i = 0; i < 2; ++i)
        async_cp16((unsigned)(size_t)&sA[buf][aRow[i] * LSTR + aCol[i]],
                   (const unsigned short*)A + (m0 + aRow[i]) * lda + kk + aCol[i]);
    } else {
#pragma unroll
      for (int i = 0; i < 4; ++i)
        raf[i] = *(const float4*)(A + (m0 + aRow[i]) * lda + kk + aCol[i]);
    }
  };
  auto storeA = [&](int buf) {
    if constexpr (!ASYNC_A) {
#pragma unroll
      for (int i = 0; i < 4; ++i) {
        uint2 pk;
        pk.x = f2bf2(raf[i].x, raf[i].y);
        pk.y = f2bf2(raf[i].z, raf[i].w);
        *(uint2*)&sA[buf][aRow[i] * LSTR + aCol[i]] = pk;
      }
    }
  };
  auto issueB = [&](int step, int buf) {
    const long kk = (long)step * BK;
    if constexpr (ASYNC_B) {
#pragma unroll
      for (int i = 0; i < 2; ++i)
        async_cp16((unsigned)(size_t)&sB[buf][bRow[i] * LSTR + bCol[i]],
                   (const unsigned short*)B + (n0 + bRow[i]) * ldb + kk + bCol[i]);
    } else if constexpr (__is_same(BT, float)) {
#pragma unroll
      for (int i = 0; i < 4; ++i)
        rbf[i] = *(const float4*)(B + (kk + bRow[i]) * ldb + n0 + bCol[i]);
    } else {
#pragma unroll
      for (int i = 0; i < 2; ++i)
        rbb[i] = *(const uint4*)((const unsigned short*)B + (kk + bRow[i]) * ldb + n0 + bCol[i]);
    }
  };
  auto storeB = [&](int buf) {
    if constexpr (BTRANS && __is_same(BT, float)) {
#pragma unroll
      for (int i = 0; i < 4; ++i) {
        sB[buf][(bCol[i] + 0) * LSTR + bRow[i]] = f2bf(rbf[i].x);
        sB[buf][(bCol[i] + 1) * LSTR + bRow[i]] = f2bf(rbf[i].y);
        sB[buf][(bCol[i] + 2) * LSTR + bRow[i]] = f2bf(rbf[i].z);
        sB[buf][(bCol[i] + 3) * LSTR + bRow[i]] = f2bf(rbf[i].w);
      }
    } else if constexpr (BTRANS) {
#pragma unroll
      for (int i = 0; i < 2; ++i) {
        union { uint4 u; unsigned short s[8]; } v; v.u = rbb[i];
#pragma unroll
        for (int j = 0; j < 8; ++j) sB[buf][(bCol[i] + j) * LSTR + bRow[i]] = v.s[j];
      }
    }
  };
  auto compute = [&](int buf) {
    FragU fa[2], fb[4];
#pragma unroll
    for (int mi = 0; mi < 2; ++mi) {
      const unsigned short* p = &sA[buf][(wm * 32 + mi * 16 + l16) * LSTR + hf * 8];
      fa[mi].h[0] = *(const v8bf*)p;        // K = hf*8 .. +7
      fa[mi].h[1] = *(const v8bf*)(p + 16); // K = 16+hf*8 .. +7
    }
#pragma unroll
    for (int ni = 0; ni < 4; ++ni) {
      const unsigned short* p = &sB[buf][(wn * 64 + ni * 16 + l16) * LSTR + hf * 8];
      fb[ni].h[0] = *(const v8bf*)p;        // K = hf*8 .. +7
      fb[ni].h[1] = *(const v8bf*)(p + 16);
    }
#pragma unroll
    for (int mi = 0; mi < 2; ++mi)
#pragma unroll
      for (int ni = 0; ni < 4; ++ni)
        acc[mi][ni] = __builtin_amdgcn_wmma_f32_16x16x32_bf16(
            false, fa[mi].v, false, fb[ni].v, (short)0, acc[mi][ni], false, false);
  };

  // ---- software pipeline: overlap next-tile loads with current WMMA ----
  const int steps = Kdim / BK;
  issueA(0, 0); issueB(0, 0);
  storeA(0);    storeB(0);
  if constexpr (ASYNC_A || ASYNC_B) wait_async0();
  __syncthreads();

  for (int s = 0; s < steps; ++s) {
    const int cb = s & 1, nb = (s + 1) & 1;
    if (s + 1 < steps) { issueA(s + 1, nb); issueB(s + 1, nb); }
    compute(cb);
    if (s + 1 < steps) {
      storeA(nb); storeB(nb);
      if constexpr (ASYNC_A || ASYNC_B) wait_async0();
      __syncthreads();
    }
  }

  // ---- epilogue: D VGPR j -> row = base + hf*8 + j, col = base + l16 ----
#pragma unroll
  for (int mi = 0; mi < 2; ++mi)
#pragma unroll
    for (int ni = 0; ni < 4; ++ni) {
      long row = m0 + wm * 32 + mi * 16 + hf * 8;
      long col = n0 + wn * 64 + ni * 16 + l16;
#pragma unroll
      for (int j = 0; j < 8; ++j) {
        float v = acc[mi][ni][j] * alpha;
        if constexpr (__is_same(CT, float)) C[(row + j) * ldc + col] = v;
        else                                C[(row + j) * ldc + col] = f2bf(v);
      }
    }
}

// Row softmax over 2048 fp32; writes bf16 probabilities overlaid in-place
// (first 4KB of each 8KB fp32 row). One wave32 per row, values in VGPRs.
__global__ __launch_bounds__(256) void softmax_rows(float* __restrict__ S)
{
  const int lane = threadIdx.x & 31;
  const int wv   = threadIdx.x >> 5;
  const long row = (long)blockIdx.x * 8 + wv;
  float* p = S + row * 2048;

  float vals[64];
  float mx = -3.4e38f;
#pragma unroll
  for (int i = 0; i < 16; ++i) {
    float4 v = *(const float4*)(p + i * 128 + lane * 4);
    vals[i * 4 + 0] = v.x; vals[i * 4 + 1] = v.y;
    vals[i * 4 + 2] = v.z; vals[i * 4 + 3] = v.w;
    mx = fmaxf(mx, fmaxf(fmaxf(v.x, v.y), fmaxf(v.z, v.w)));
  }
#pragma unroll
  for (int off = 16; off > 0; off >>= 1) mx = fmaxf(mx, __shfl_xor(mx, off, 32));

  float sum = 0.0f;
#pragma unroll
  for (int i = 0; i < 64; ++i) { vals[i] = __expf(vals[i] - mx); sum += vals[i]; }
#pragma unroll
  for (int off = 16; off > 0; off >>= 1) sum += __shfl_xor(sum, off, 32);
  const float inv = 1.0f / sum;

  unsigned short* q = (unsigned short*)p;   // bf16 overlay, same element indices
#pragma unroll
  for (int i = 0; i < 16; ++i) {
    uint2 pk;
    pk.x = f2bf2(vals[i * 4 + 0] * inv, vals[i * 4 + 1] * inv);
    pk.y = f2bf2(vals[i * 4 + 2] * inv, vals[i * 4 + 3] * inv);
    *(uint2*)(q + i * 128 + lane * 4) = pk;
  }
}

extern "C" void kernel_launch(void* const* d_in, const int* in_sizes, int n_in,
                              void* d_out, int out_size, void* d_ws, size_t ws_size,
                              hipStream_t stream)
{
  const float* x    = (const float*)d_in[0];   // [4][2048][1024]
  const float* kern = (const float*)d_in[1];   // [3][1024][1024]  (0:K 1:Q 2:V)
  float* out = (float*)d_out;                  // [4][2048][1024]

  const long QKV = 8192L * 1024;               // elems per projected matrix
  unsigned short* Kb = (unsigned short*)d_ws;  // bf16
  unsigned short* Qb = Kb + QKV;
  unsigned short* Vb = Qb + QKV;
  float* Sc = (float*)((char*)d_ws + 3 * QKV * 2);  // fp32 scores [4][2048][2048]

  dim3 blk(256);

  // 1) K/Q/V = X @ kernel[z]
  gemm_wmma<float, float, unsigned short, true>
      <<<dim3(64, 8, 3), blk, 0, stream>>>(
          x, 1024, 0,
          kern, 1024, 1024L * 1024,
          Kb, 1024, QKV, 1024, 1.0f);

  // 2) scores_b = (Q_b @ K_b^T) / 32 ; K row-major [key][d] => async direct copy
  gemm_wmma<unsigned short, unsigned short, float, false>
      <<<dim3(16, 16, 4), blk, 0, stream>>>(
          Qb, 1024, 2048L * 1024,
          Kb, 1024, 2048L * 1024,
          Sc, 2048, 2048L * 2048, 1024, 0.03125f);

  // 3) softmax over each of the 8192 rows; P bf16 overlaid (row stride 4096 bf16)
  softmax_rows<<<dim3(1024), blk, 0, stream>>>(Sc);

  // 4) out_b = P_b @ V_b
  gemm_wmma<unsigned short, unsigned short, float, true>
      <<<dim3(16, 8, 4), blk, 0, stream>>>(
          (const unsigned short*)Sc, 4096, 2048L * 4096,
          Vb, 1024, 2048L * 1024,
          out, 1024, 2048L * 1024, 2048, 1.0f);
}
